// Edgewise_27547920236674
// MI455X (gfx1250) — matrix operand order repaired and checked by reading, not compiled, not run on hardware
//
#include <hip/hip_runtime.h>

typedef __bf16 bf16_t;
typedef __attribute__((ext_vector_type(16))) __bf16 v16bf;
typedef __attribute__((ext_vector_type(8)))  __bf16 v8bf;
typedef __attribute__((ext_vector_type(8)))  float  v8f;
typedef __attribute__((ext_vector_type(4)))  int    v4i;

#define ASYNC_LDS_OK (__has_builtin(__builtin_amdgcn_global_load_async_to_lds_b128) && \
                      __has_builtin(__builtin_amdgcn_s_wait_asynccnt))

typedef __attribute__((address_space(1))) v4i* g4i_ptr;   // global v4i*
typedef __attribute__((address_space(3))) v4i* l4i_ptr;   // LDS v4i*

__device__ __forceinline__ float sigmoidf_(float x) { return 1.0f / (1.0f + __expf(-x)); }

// x_rad offset for msg row mc (channel c added by caller)
__device__ __forceinline__ int rad_off(int mc) {
    if (mc < 7)  return mc * 256;
    if (mc < 13) return 1792 + (mc - 7)  * 256;
    if (mc < 19) return 1792 + (mc - 13) * 256;
    if (mc < 24) return 3328 + (mc - 19) * 256;
    return 3328 + (mc - 24) * 256;
}

// -------- vectorized WMMA fragment loaders (two 16B LDS loads each) --------
__device__ __forceinline__ v16bf frag16(const bf16_t* p0, const bf16_t* p1) {
    v8bf lo = *(const v8bf*)p0;
    v8bf hi = *(const v8bf*)p1;
    return __builtin_shufflevector(lo, hi, 0, 1, 2, 3, 4, 5, 6, 7, 8, 9, 10, 11, 12, 13, 14, 15);
}
// A fragment: row-major 16x32 tile at `tile`, row stride lda elems (lda % 8 == 0)
__device__ __forceinline__ v16bf loadA(const bf16_t* tile, int lane, int lda) {
    int m = lane & 15;
    int kb = (lane < 16) ? 0 : 8;
    const bf16_t* p = tile + m * lda + kb;
    return frag16(p, p + 16);
}
// B fragment: TRANSPOSED tile [n][k] at `tileT`, col stride ldk elems (ldk % 8 == 0)
__device__ __forceinline__ v16bf loadB(const bf16_t* tileT, int lane, int ldk) {
    int n = lane & 15;
    int kb = (lane < 16) ? 0 : 8;
    const bf16_t* p = tileT + n * ldk + kb;
    return frag16(p, p + 16);
}

// ---------------- weight conversion f32 -> bf16 ----------------
__global__ void cvt_bf16_kernel(const float* __restrict__ src, bf16_t* __restrict__ dst, long n) {
    long i = (long)blockIdx.x * blockDim.x + threadIdx.x;
    long stride = (long)gridDim.x * blockDim.x;
    for (; i < n; i += stride) dst[i] = (bf16_t)src[i];
}

// ---------------- generic bf16 WMMA GEMM ----------------
// C[M,N] = A[M,K] @ B[K,N] (+bias). A rows addressed via:
//   off = (row / rowsPerEdge)*edgeStride + baseOff + (row % rowsPerEdge)*rStride
// 256 threads (8 waves), 64x64 C tile, K-step 32 (all K used are multiples of 32).
// B tile staged transposed in LDS; bf16 A tiles use async global->LDS copy when available.
template <typename TA>
__global__ void gemm_ws(const TA* __restrict__ A, const bf16_t* __restrict__ B,
                        const float* __restrict__ bias, float* __restrict__ C,
                        int M, int N, int K,
                        int rowsPerEdge, long edgeStride, long baseOff, long rStride) {
    __shared__ bf16_t lA[64 * 40];    // 64 rows x 32 K (stride 40)
    __shared__ bf16_t lBt[64 * 40];   // 64 n   x 32 K (stride 40), transposed
    const int t = threadIdx.x;
    const int lane = t & 31;
    const int w = t >> 5;                 // 0..7
    const int mBase = blockIdx.x * 64;
    const int nBase = blockIdx.y * 64;
    const int mSub = w & 3;               // 0..3
    const int nSub0 = (w >> 2) * 2;       // 0 or 2

    v8f acc0 = {}; v8f acc1 = {};
    for (int kt = 0; kt < K; kt += 32) {
        bool usedAsync = false;
#if ASYNC_LDS_OK
        if constexpr (__is_same(TA, bf16_t)) {
            // A tile 64x32 bf16 = 4 KB: one b128 async copy per thread, no VGPR round-trip.
            int r = t >> 2, k0 = (t & 3) * 8;
            int gr = mBase + r;
            if (gr >= M) gr = M - 1;      // clamp: those C rows are never stored
            long off = (long)(gr / rowsPerEdge) * edgeStride + baseOff +
                       (long)(gr % rowsPerEdge) * rStride;
            bf16_t* gp = const_cast<bf16_t*>(A + off + kt + k0);
            bf16_t* lp = lA + r * 40 + k0;
            __builtin_amdgcn_global_load_async_to_lds_b128(
                (g4i_ptr)(void*)gp, (l4i_ptr)(void*)lp, 0, 0);
            usedAsync = true;
        }
#endif
        if (!usedAsync) {
            #pragma unroll
            for (int i = 0; i < 8; ++i) { // A tile: 64x32 (sync path / f32 source)
                int idx = t + i * 256;
                int r = idx >> 5, k = idx & 31;
                float v = 0.0f;
                int gr = mBase + r, gk = kt + k;
                if (gr < M && gk < K) {
                    long off = (long)(gr / rowsPerEdge) * edgeStride + baseOff +
                               (long)(gr % rowsPerEdge) * rStride;
                    v = (float)A[off + gk];
                }
                lA[r * 40 + k] = (bf16_t)v;
            }
        }
        #pragma unroll
        for (int i = 0; i < 8; ++i) {     // B tile: 32x64 -> transposed store
            int idx = t + i * 256;
            int kr = idx >> 6, n = idx & 63;
            float v = 0.0f;
            int gk = kt + kr, gn = nBase + n;
            if (gk < K && gn < N) v = (float)B[(long)gk * N + gn];
            lBt[n * 40 + kr] = (bf16_t)v;
        }
#if ASYNC_LDS_OK
        if constexpr (__is_same(TA, bf16_t)) __builtin_amdgcn_s_wait_asynccnt(0);
#endif
        __syncthreads();
        v16bf a = loadA(lA + mSub * 16 * 40, lane, 40);
        v16bf b0 = loadB(lBt + (nSub0 + 0) * 16 * 40, lane, 40);
        v16bf b1 = loadB(lBt + (nSub0 + 1) * 16 * 40, lane, 40);
        acc0 = __builtin_amdgcn_wmma_f32_16x16x32_bf16(false, a, false, b0, (short)0, acc0, false, false);
        acc1 = __builtin_amdgcn_wmma_f32_16x16x32_bf16(false, a, false, b1, (short)0, acc1, false, false);
        __syncthreads();
    }
    int colL = lane & 15;
    int rowHi = (lane < 16) ? 0 : 8;
    #pragma unroll
    for (int r = 0; r < 8; ++r) {
        int row = mBase + mSub * 16 + rowHi + r;
        if (row >= M) continue;
        #pragma unroll
        for (int s = 0; s < 2; ++s) {
            int col = nBase + (nSub0 + s) * 16 + colL;
            if (col >= N) continue;
            float v = (s == 0) ? acc0[r] : acc1[r];
            if (bias) v += bias[col];
            C[(long)row * N + col] = v;
        }
    }
}

// ---------------- LayerNorm + SiLU -> bf16 (rows of 256) ----------------
__global__ void ln_silu_kernel(const float* __restrict__ H, const float* __restrict__ s,
                               const float* __restrict__ b, bf16_t* __restrict__ O) {
    __shared__ float red[256];
    long row = blockIdx.x;
    int t = threadIdx.x;
    float v = H[row * 256 + t];
    red[t] = v; __syncthreads();
    for (int o = 128; o > 0; o >>= 1) { if (t < o) red[t] += red[t + o]; __syncthreads(); }
    float mu = red[0] * (1.0f / 256.0f); __syncthreads();
    float d = v - mu;
    red[t] = d * d; __syncthreads();
    for (int o = 128; o > 0; o >>= 1) { if (t < o) red[t] += red[t + o]; __syncthreads(); }
    float var = red[0] * (1.0f / 256.0f);
    float h = d * rsqrtf(var + 1e-5f) * s[t] + b[t];
    float out = h * sigmoidf_(h);
    O[row * 256 + t] = (bf16_t)out;
}

// ---------------- per-edge: msg = wig @ concat(x[src],x[dst]), scaled by x_rad ----------------
// one block (256 thr / 8 waves) per edge; C tile 32x256 via WMMA
__global__ void msg_kernel(const float* __restrict__ x, const int* __restrict__ eSrc,
                           const int* __restrict__ eDst, const float* __restrict__ wig,
                           const float* __restrict__ xrad, bf16_t* __restrict__ msgb) {
    __shared__ bf16_t lA[32 * 72];     // wig 29x49 padded to 32x64 (stride 72)
    __shared__ bf16_t lBt[256 * 72];   // xcat 49x256 padded to 64x256, stored [n][k]
    const int t = threadIdx.x;
    const int lane = t & 31;
    const int w = t >> 5;              // 0..7
    const long e = blockIdx.x;
    const int src = eSrc[e], dst = eDst[e];

    const float* wg = wig + e * (29L * 49);
    #pragma unroll
    for (int i = 0; i < 8; ++i) {
        int idx = t + i * 256;         // 0..2047 (32x64)
        int r = idx >> 6, k = idx & 63;
        float v = (r < 29 && k < 49) ? wg[r * 49 + k] : 0.0f;
        lA[r * 72 + k] = (bf16_t)v;
    }
    const float* xs = x + (long)src * (49 * 128);
    const float* xd = x + (long)dst * (49 * 128);
    for (int i = 0; i < 64; ++i) {
        int idx = t + i * 256;         // 0..16383 (64x256)
        int r = idx >> 8, c = idx & 255;
        float v = 0.0f;
        if (r < 49) v = (c < 128) ? xs[r * 128 + c] : xd[r * 128 + (c - 128)];
        lBt[c * 72 + r] = (bf16_t)v;   // transposed store
    }
    __syncthreads();

    const int mSub = w & 1;
    const int nSub0 = (w >> 1) * 4;    // 0,4,8,12
    v8f acc[4] = {};
    #pragma unroll
    for (int kt = 0; kt < 64; kt += 32) {
        v16bf a = loadA(lA + mSub * 16 * 72 + kt, lane, 72);
        #pragma unroll
        for (int s = 0; s < 4; ++s) {
            v16bf b = loadB(lBt + (nSub0 + s) * 16 * 72 + kt, lane, 72);
            acc[s] = __builtin_amdgcn_wmma_f32_16x16x32_bf16(false, a, false, b, (short)0, acc[s], false, false);
        }
    }
    const float* xr = xrad + e * 4608;
    bf16_t* mo = msgb + e * 7424;
    int colL = lane & 15;
    int rowHi = (lane < 16) ? 0 : 8;
    #pragma unroll
    for (int s = 0; s < 4; ++s) {
        int n0 = (nSub0 + s) * 16 + colL;
        #pragma unroll
        for (int r = 0; r < 8; ++r) {
            int mc = mSub * 16 + rowHi + r;
            if (mc >= 29) continue;
            float v = acc[s][r] * xr[rad_off(mc) + n0];
            mo[mc * 256 + n0] = (bf16_t)v;
        }
    }
}

// ---------------- complex combine + gating -> conv2 input (bf16) ----------------
__global__ void combine_gate_kernel(const float* __restrict__ y0, const float* __restrict__ y1,
                                    const float* __restrict__ y2, bf16_t* __restrict__ z) {
    long e = blockIdx.x;
    const float* Y0 = y0 + e * 1664;
    const float* Y1 = y1 + e * 2 * 1536;
    const float* Y2 = y2 + e * 2 * 1280;
    bf16_t* Z = z + e * 3712;
    for (int idx = threadIdx.x; idx < 3712; idx += blockDim.x) {
        int row = idx >> 7, c = idx & 127;
        float v; int g = -1;
        if (row < 7) {
            v = Y0[768 + row * 128 + c];
            if (row >= 1) g = row - 1;
        } else if (row < 13) { int j = row - 7;  v = Y1[j * 128 + c] - Y1[1536 + 768 + j * 128 + c]; g = j; }
        else if (row < 19)   { int j = row - 13; v = Y1[1536 + j * 128 + c] + Y1[768 + j * 128 + c]; g = j; }
        else if (row < 24)   { int j = row - 19; v = Y2[j * 128 + c] - Y2[1280 + 640 + j * 128 + c]; g = 1 + j; }
        else                 { int j = row - 24; v = Y2[1280 + j * 128 + c] + Y2[640 + j * 128 + c]; g = 1 + j; }
        if (g < 0) v = v * sigmoidf_(v);                    // silu on l=0 scalars
        else       v = v * sigmoidf_(Y0[g * 128 + c]);      // gate
        Z[idx] = (bf16_t)v;
    }
}

// ---------------- conv2 complex combine + envelope -> bf16 msg2 ----------------
__global__ void combine2_env_kernel(const float* __restrict__ u0, const float* __restrict__ u1,
                                    const float* __restrict__ u2, const float* __restrict__ env,
                                    bf16_t* __restrict__ m2) {
    long e = blockIdx.x;
    float ev = env[e];
    const float* U0 = u0 + e * 896;
    const float* U1 = u1 + e * 2 * 1536;
    const float* U2 = u2 + e * 2 * 1280;
    bf16_t* M = m2 + e * 3712;
    for (int idx = threadIdx.x; idx < 3712; idx += blockDim.x) {
        int row = idx >> 7, c = idx & 127;
        float v;
        if (row < 7)        v = U0[row * 128 + c];
        else if (row < 13) { int j = row - 7;  v = U1[j * 128 + c] - U1[1536 + 768 + j * 128 + c]; }
        else if (row < 19) { int j = row - 13; v = U1[1536 + j * 128 + c] + U1[768 + j * 128 + c]; }
        else if (row < 24) { int j = row - 19; v = U2[j * 128 + c] - U2[1280 + 640 + j * 128 + c]; }
        else               { int j = row - 24; v = U2[1280 + j * 128 + c] + U2[640 + j * 128 + c]; }
        M[idx] = (bf16_t)(v * ev);
    }
}

// ---------------- per-edge: out[dst] += wig_inv @ msg2  (WMMA + f32 atomics) ----------------
__global__ void scatter_kernel(const float* __restrict__ wigInv, const bf16_t* __restrict__ m2,
                               const int* __restrict__ eDst, float* __restrict__ out) {
    __shared__ bf16_t lA[64 * 40];     // wig_inv 49x29 padded to 64x32 (stride 40)
    __shared__ bf16_t lBt[128 * 40];   // msg2 29x128 padded to 32x128, stored [n][k]
    const int t = threadIdx.x;
    const int lane = t & 31;
    const int w = t >> 5;
    const long e = blockIdx.x;
    const float* Wi = wigInv + e * (49L * 29);
    const bf16_t* M = m2 + e * 3712;
    #pragma unroll
    for (int i = 0; i < 8; ++i) {
        int idx = t + i * 256;         // 0..2047 (64x32)
        int r = idx >> 5, k = idx & 31;
        lA[r * 40 + k] = (r < 49 && k < 29) ? (bf16_t)Wi[r * 29 + k] : (bf16_t)0.0f;
    }
    #pragma unroll
    for (int i = 0; i < 16; ++i) {
        int idx = t + i * 256;         // 0..4095 (32x128)
        int k = idx >> 7, n = idx & 127;
        lBt[n * 40 + k] = (k < 29) ? M[idx] : (bf16_t)0.0f;   // transposed store
    }
    __syncthreads();
    const int mSub = w >> 1;           // 0..3
    const int nSub0 = (w & 1) * 4;     // 0 or 4
    v8f acc[4] = {};
    v16bf a = loadA(lA + mSub * 16 * 40, lane, 40);
    #pragma unroll
    for (int s = 0; s < 4; ++s) {
        v16bf b = loadB(lBt + (nSub0 + s) * 16 * 40, lane, 40);
        acc[s] = __builtin_amdgcn_wmma_f32_16x16x32_bf16(false, a, false, b, (short)0, acc[s], false, false);
    }
    const int dst = eDst[e];
    float* O = out + (long)dst * (49 * 128);
    int colL = lane & 15;
    int rowHi = (lane < 16) ? 0 : 8;
    #pragma unroll
    for (int s = 0; s < 4; ++s) {
        int col = (nSub0 + s) * 16 + colL;
        #pragma unroll
        for (int r = 0; r < 8; ++r) {
            int row = mSub * 16 + rowHi + r;
            if (row < 49) atomicAdd(&O[row * 128 + col], acc[s][r]);
        }
    }
}

// =======================================================================
extern "C" void kernel_launch(void* const* d_in, const int* in_sizes, int n_in,
                              void* d_out, int out_size, void* d_ws, size_t ws_size,
                              hipStream_t stream) {
    const float* x        = (const float*)d_in[0];
    const float* x_edge   = (const float*)d_in[1];
    const int*   eidx     = (const int*)d_in[3];
    const float* wig      = (const float*)d_in[4];
    const float* wigInv   = (const float*)d_in[5];
    const float* env      = (const float*)d_in[6];
    const float* rad_w0   = (const float*)d_in[7];
    const float* rad_b0   = (const float*)d_in[8];
    const float* rad_s0   = (const float*)d_in[9];
    const float* rad_bb0  = (const float*)d_in[10];
    const float* rad_w1   = (const float*)d_in[11];
    const float* rad_b1   = (const float*)d_in[12];
    const float* rad_s1   = (const float*)d_in[13];
    const float* rad_bb1  = (const float*)d_in[14];
    const float* rad_w2   = (const float*)d_in[15];
    const float* rad_b2   = (const float*)d_in[16];
    const float* c1w0     = (const float*)d_in[17];
    const float* c1b0     = (const float*)d_in[18];
    const float* c1w1     = (const float*)d_in[19];
    const float* c1w2     = (const float*)d_in[20];
    const float* c2w0     = (const float*)d_in[21];
    const float* c2b0     = (const float*)d_in[22];
    const float* c2w1     = (const float*)d_in[23];
    const float* c2w2     = (const float*)d_in[24];
    float* out = (float*)d_out;

    const long nEdges = in_sizes[6];                 // edge_envelope count
    (void)n_in;

    (void)hipMemsetAsync(d_out, 0, (size_t)out_size * sizeof(float), stream);

    char* ws = (char*)d_ws;
    auto alignup = [](size_t v) { return (v + 255) & ~(size_t)255; };
    size_t off = 0;

    // ---- bf16 weight copies ----
    struct WEntry { const float* s; long n; };
    WEntry We[9] = {
        { rad_w0, 128L * 256 }, { rad_w1, 256L * 256 }, { rad_w2, 256L * 4608 },
        { c1w0, 1792L * 1664 }, { c1w1, 1536L * 1536 }, { c1w2, 1280L * 1280 },
        { c2w0, 896L * 896 },   { c2w1, 768L * 1536 },  { c2w2, 640L * 1280 }
    };
    bf16_t* Wb[9];
    for (int i = 0; i < 9; ++i) { Wb[i] = (bf16_t*)(ws + off); off = alignup(off + (size_t)We[i].n * 2); }
    for (int i = 0; i < 9; ++i) {
        long blocks = (We[i].n + 255) / 256;
        if (blocks > 2048) blocks = 2048;
        cvt_bf16_kernel<<<(int)blocks, 256, 0, stream>>>(We[i].s, Wb[i], We[i].n);
    }
    size_t weightsEnd = off;

    // ---- per-edge chunk sizing from ws_size ----
    const size_t perEdge = 106496;                   // bytes of intermediates per edge
    size_t avail = (ws_size > weightsEnd + 8192) ? (ws_size - weightsEnd - 8192) : perEdge;
    long CE = (long)(avail / perEdge);
    if (CE < 1) CE = 1;
    if (CE > nEdges) CE = nEdges;

    size_t o = weightsEnd;
    auto take = [&](size_t bytes) { void* p = ws + o; o = alignup(o + bytes); return p; };
    float*  h0   = (float*) take((size_t)CE * 256 * 4);
    bf16_t* h0b  = (bf16_t*)take((size_t)CE * 256 * 2);
    float*  h1   = (float*) take((size_t)CE * 256 * 4);
    bf16_t* h1b  = (bf16_t*)take((size_t)CE * 256 * 2);
    float*  xrad = (float*) take((size_t)CE * 4608 * 4);
    bf16_t* msgb = (bf16_t*)take((size_t)CE * 7424 * 2);
    float*  y0   = (float*) take((size_t)CE * 1664 * 4);
    float*  y1   = (float*) take((size_t)CE * 3072 * 4);
    float*  y2   = (float*) take((size_t)CE * 2560 * 4);
    bf16_t* z    = (bf16_t*)take((size_t)CE * 3712 * 2);
    float*  u0   = (float*) take((size_t)CE * 896 * 4);
    float*  u1   = (float*) take((size_t)CE * 3072 * 4);
    float*  u2   = (float*) take((size_t)CE * 2560 * 4);
    bf16_t* m2   = (bf16_t*)take((size_t)CE * 3712 * 2);

    dim3 blk(256);
    for (long e0 = 0; e0 < nEdges; e0 += CE) {
        long ec = (nEdges - e0 < CE) ? (nEdges - e0) : CE;
        int gM  = (int)((ec + 63) / 64);
        int gM2 = (int)((2 * ec + 63) / 64);

        // radial MLP
        gemm_ws<float><<<dim3(gM, 256 / 64), blk, 0, stream>>>(
            x_edge + e0 * 128, Wb[0], rad_b0, h0, (int)ec, 256, 128, 1, 128, 0, 0);
        ln_silu_kernel<<<(int)ec, 256, 0, stream>>>(h0, rad_s0, rad_bb0, h0b);
        gemm_ws<bf16_t><<<dim3(gM, 256 / 64), blk, 0, stream>>>(
            h0b, Wb[1], rad_b1, h1, (int)ec, 256, 256, 1, 256, 0, 0);
        ln_silu_kernel<<<(int)ec, 256, 0, stream>>>(h1, rad_s1, rad_bb1, h1b);
        gemm_ws<bf16_t><<<dim3(gM, 4608 / 64), blk, 0, stream>>>(
            h1b, Wb[2], rad_b2, xrad, (int)ec, 4608, 256, 1, 256, 0, 0);

        // Wigner rotation + radial scaling
        msg_kernel<<<(int)ec, 256, 0, stream>>>(
            x, eidx + e0, eidx + nEdges + e0, wig + e0 * (29L * 49), xrad, msgb);

        // SO(2) conv 1
        gemm_ws<bf16_t><<<dim3(gM, 1664 / 64), blk, 0, stream>>>(
            msgb, Wb[3], c1b0, y0, (int)ec, 1664, 1792, 1, 7424, 0, 0);
        gemm_ws<bf16_t><<<dim3(gM2, 1536 / 64), blk, 0, stream>>>(
            msgb, Wb[4], nullptr, y1, (int)(2 * ec), 1536, 1536, 2, 7424, 1792, 1536);
        gemm_ws<bf16_t><<<dim3(gM2, 1280 / 64), blk, 0, stream>>>(
            msgb, Wb[5], nullptr, y2, (int)(2 * ec), 1280, 1280, 2, 7424, 4864, 1280);
        combine_gate_kernel<<<(int)ec, 256, 0, stream>>>(y0, y1, y2, z);

        // SO(2) conv 2
        gemm_ws<bf16_t><<<dim3(gM, 896 / 64), blk, 0, stream>>>(
            z, Wb[6], c2b0, u0, (int)ec, 896, 896, 1, 3712, 0, 0);
        gemm_ws<bf16_t><<<dim3(gM2, 1536 / 64), blk, 0, stream>>>(
            z, Wb[7], nullptr, u1, (int)(2 * ec), 1536, 768, 2, 3712, 896, 768);
        gemm_ws<bf16_t><<<dim3(gM2, 1280 / 64), blk, 0, stream>>>(
            z, Wb[8], nullptr, u2, (int)(2 * ec), 1280, 640, 2, 3712, 2432, 640);
        combine2_env_kernel<<<(int)ec, 256, 0, stream>>>(u0, u1, u2, env + e0, m2);

        // inverse Wigner + segment-sum scatter
        scatter_kernel<<<(int)ec, 256, 0, stream>>>(
            wigInv + e0 * (49L * 29), m2, eidx + nEdges + e0, out);
    }
}
